// PadimModel_11201274708746
// MI455X (gfx1250) — matrix operand order repaired and checked by reading, not compiled, not run on hardware
//
#include <hip/hip_runtime.h>
#include <hip/hip_bf16.h>
#include <stdint.h>

// Problem constants (from reference)
#define B_SZ    32
#define C_CMB   448
#define HW_IN   32
#define SP_IN   1024     // 32*32
#define HW_FT   64
#define NPOS    4096
#define DPROJ   100
#define DPAD    112      // proj row width  (7 * 16 WMMA N-tiles)
#define DPAD2   128      // diff row width  (4 * 32 WMMA K-steps), pad zeros
#define OUT_HW  256

typedef __attribute__((ext_vector_type(16))) __bf16 v16bf;
typedef __attribute__((ext_vector_type(2)))  __bf16 v2bf;
typedef __attribute__((ext_vector_type(8)))  float  v8f;

union BF16x16 { v16bf v; unsigned int u[8]; };

// f32 pair -> packed bf16 dword (lo | hi<<16) in ONE VALU op.
// Prefer the native packed convert (RNE); fall back to v_perm_b32 truncation.
#if defined(__has_builtin)
#if __has_builtin(__builtin_amdgcn_cvt_pk_bf16_f32)
#define HAVE_CVT_PK_BF16 1
#endif
#endif
__device__ __forceinline__ unsigned int bf16pack2(float lo, float hi) {
#ifdef HAVE_CVT_PK_BF16
  v2bf r = __builtin_amdgcn_cvt_pk_bf16_f32(lo, hi);
  return __builtin_bit_cast(unsigned int, r);
#else
  // result bytes: {hi[3],hi[2],lo[3],lo[2]}  (sel 0-3 -> src1, 4-7 -> src0)
  return __builtin_amdgcn_perm(__float_as_uint(hi), __float_as_uint(lo),
                               0x07060302u);
#endif
}
__device__ __forceinline__ unsigned short bf16s(float x) {
  return (unsigned short)(bf16pack2(x, 0.f) & 0xFFFFu);
}

__device__ __forceinline__ v8f wmma_bf16(const v16bf& A, const v16bf& B, v8f C) {
  return __builtin_amdgcn_wmma_f32_16x16x32_bf16(false, A, false, B,
                                                 (short)0, C, false, false);
}

// ---------------------------------------------------------------------------
// Kernel B0: adjT[n][d] = bias[d] - mean[d][n]   (fold bias into -mean), pad 0
// ---------------------------------------------------------------------------
__global__ void __launch_bounds__(128)
make_adj(const float* __restrict__ mean, const float* __restrict__ bias,
         float* __restrict__ adjT) {
  const int n = blockIdx.x;
  const int d = threadIdx.x;
  float v = 0.f;
  if (d < DPROJ) v = bias[d] - mean[(size_t)d * NPOS + n];
  adjT[(size_t)n * DPAD2 + d] = v;
}

// ---------------------------------------------------------------------------
// Kernel A: projection GEMM at 32x32 resolution (projection commutes with
// bilinear interp -> 4x fewer FLOPs, no 235MB upsampled intermediate).
// Per (b, m-tile): M=16 spatial rows, N=112 (7 tiles), K=448 (14x32).
// ---------------------------------------------------------------------------
__global__ void __launch_bounds__(32)
proj_gemm(const float* __restrict__ comb, const float* __restrict__ W,
          float* __restrict__ proj) {
  const int mt   = blockIdx.x;        // 0..63
  const int b    = blockIdx.y;        // 0..31
  const int lane = threadIdx.x;
  const int half = lane >> 4;
  const int l15  = lane & 15;
  const int m    = mt * 16 + l15;     // spatial index (A-matrix row M)

  const float* src = comb + (size_t)b * C_CMB * SP_IN;

  const v8f zero = {0.f,0.f,0.f,0.f,0.f,0.f,0.f,0.f};
  v8f acc[7];
#pragma unroll
  for (int t = 0; t < 7; ++t) acc[t] = zero;

  for (int ks = 0; ks < 14; ++ks) {
    const int c0 = ks * 32;
    // A fragment: contiguous k order (same as B). lanes<16: K=0..15, lanes>=16: K=16..31
    BF16x16 af;
#pragma unroll
    for (int j = 0; j < 8; ++j) {
      const int c = c0 + half * 16 + 2 * j;
      const float f0 = src[(size_t)c * SP_IN + m];
      const float f1 = src[(size_t)(c + 1) * SP_IN + m];
      af.u[j] = bf16pack2(f0, f1);
    }
#pragma unroll
    for (int nt = 0; nt < 7; ++nt) {
      const int  d   = nt * 16 + l15;           // output channel (N)
      const bool dok = d < DPROJ;               // folds true for nt<6
      const int  dc  = dok ? d : 0;
      // B layout: lanes<16 hold K=0..15, lanes>=16 hold K=16..31 (contig in c)
      const float* wr = W + (size_t)dc * C_CMB + c0 + half * 16;
      float4 q0 = *(const float4*)(wr + 0);
      float4 q1 = *(const float4*)(wr + 4);
      float4 q2 = *(const float4*)(wr + 8);
      float4 q3 = *(const float4*)(wr + 12);
      if (!dok) { q0 = make_float4(0,0,0,0); q1 = q0; q2 = q0; q3 = q0; }
      BF16x16 bf;
      bf.u[0] = bf16pack2(q0.x, q0.y);  bf.u[1] = bf16pack2(q0.z, q0.w);
      bf.u[2] = bf16pack2(q1.x, q1.y);  bf.u[3] = bf16pack2(q1.z, q1.w);
      bf.u[4] = bf16pack2(q2.x, q2.y);  bf.u[5] = bf16pack2(q2.z, q2.w);
      bf.u[6] = bf16pack2(q3.x, q3.y);  bf.u[7] = bf16pack2(q3.z, q3.w);
      acc[nt] = wmma_bf16(af.v, bf.v, acc[nt]);
    }
  }
  // C/D layout store: VGPR r -> M = r + 8*half, N = l15. d>=100 lanes hold 0.
  float* ob = proj + (size_t)b * SP_IN * DPAD;
#pragma unroll
  for (int nt = 0; nt < 7; ++nt) {
#pragma unroll
    for (int r = 0; r < 8; ++r) {
      const int mo = mt * 16 + r + 8 * half;
      const int d  = nt * 16 + l15;
      ob[(size_t)mo * DPAD + d] = acc[nt][r];
    }
  }
}

// ---------------------------------------------------------------------------
// Kernel B: bilinear 32->64 (half-pixel, clamped) of projected features,
// add (bias - mean), emit bf16 diff rows [n][b][128] (zero padded d>=100).
// ---------------------------------------------------------------------------
__global__ void __launch_bounds__(128)
upsample_diff(const float* __restrict__ proj, const float* __restrict__ adjT,
              unsigned short* __restrict__ diff) {
  const int n = blockIdx.x;
  const int b = blockIdx.y;
  const int d = threadIdx.x;
  const int y2 = n >> 6, x2 = n & 63;
  const float sy = y2 * 0.5f - 0.25f;
  const float sx = x2 * 0.5f - 0.25f;
  int y0 = (int)floorf(sy); const float fy = sy - (float)y0;
  int x0 = (int)floorf(sx); const float fx = sx - (float)x0;
  const int y1 = min(y0 + 1, HW_IN - 1); y0 = max(y0, 0);
  const int x1 = min(x0 + 1, HW_IN - 1); x0 = max(x0, 0);
  float v = 0.f;
  if (d < DPROJ) {
    const float* pbm = proj + (size_t)b * SP_IN * DPAD;
    const float s00 = pbm[(size_t)(y0 * HW_IN + x0) * DPAD + d];
    const float s01 = pbm[(size_t)(y0 * HW_IN + x1) * DPAD + d];
    const float s10 = pbm[(size_t)(y1 * HW_IN + x0) * DPAD + d];
    const float s11 = pbm[(size_t)(y1 * HW_IN + x1) * DPAD + d];
    const float top = s00 + fx * (s01 - s00);
    const float bot = s10 + fx * (s11 - s10);
    v = top + fy * (bot - top) + adjT[(size_t)n * DPAD2 + d];
  }
  diff[((size_t)n * B_SZ + b) * DPAD2 + d] = bf16s(v);
}

// ---------------------------------------------------------------------------
// Kernel C: per-pixel Mahalanobis via WMMA. One N-tile (16 Sigma columns):
// acc = D * Sigma[:, tile] over K=128, then part += T .* D (C-layout dot).
// NT is compile-time so all 100-bounds masking folds away except the
// last K-step (ks==3) and last N-tile (NT==6).
// ---------------------------------------------------------------------------
template <int NT>
__device__ __forceinline__ void maha_tile(
    const float* __restrict__ S, const unsigned short* __restrict__ drows,
    const BF16x16 (&a)[2][4], const int half16, const int l15,
    float (&part)[2][8]) {
  const int  dcol = NT * 16 + l15;                 // Sigma column (N)
  const bool dok  = (NT < 6) ? true : (dcol < DPROJ);
  const int  dcl  = dok ? dcol : 0;
  const v8f zero = {0.f,0.f,0.f,0.f,0.f,0.f,0.f,0.f};
  v8f acc0 = zero, acc1 = zero;
#pragma unroll
  for (int ks = 0; ks < 4; ++ks) {
    BF16x16 bf;
#pragma unroll
    for (int j = 0; j < 8; ++j) {
      const int cbase = ks * 32 + 2 * j;           // compile-time
      if (cbase + 17 < DPROJ) {                    // fully in-bounds K pair
        float f0 = S[(size_t)(cbase + half16) * DPROJ + dcl];
        float f1 = S[(size_t)(cbase + half16 + 1) * DPROJ + dcl];
        if (NT == 6) { f0 = dok ? f0 : 0.f; f1 = dok ? f1 : 0.f; }
        bf.u[j] = bf16pack2(f0, f1);
      } else {                                     // K edge (ks==3 only)
        const int  c  = cbase + half16;
        const bool k0 = (c < DPROJ);
        const bool k1 = (c + 1 < DPROJ);
        float f0 = S[(size_t)(k0 ? c : 0) * DPROJ + dcl];
        float f1 = S[(size_t)(k1 ? (c + 1) : 0) * DPROJ + dcl];
        f0 = (k0 && dok) ? f0 : 0.f;
        f1 = (k1 && dok) ? f1 : 0.f;
        bf.u[j] = bf16pack2(f0, f1);
      }
    }
    acc0 = wmma_bf16(a[0][ks].v, bf.v, acc0);
    acc1 = wmma_bf16(a[1][ks].v, bf.v, acc1);
  }
  // row-wise dot: T (C-layout: M = r + 8*half, N = l15) * D (same layout).
  // diff rows are zero-padded to 128 -> no masking needed (dcol <= 111).
#pragma unroll
  for (int r = 0; r < 8; ++r) {
    const int b0 = r + (half16 >> 1);              // r + 8*half
    const float d0v = __uint_as_float(
        (unsigned int)drows[(size_t)b0 * DPAD2 + dcol] << 16);
    const float d1v = __uint_as_float(
        (unsigned int)drows[(size_t)(16 + b0) * DPAD2 + dcol] << 16);
    part[0][r] += acc0[r] * d0v;
    part[1][r] += acc1[r] * d1v;
  }
}

__global__ void __launch_bounds__(32)
mahalanobis(const unsigned short* __restrict__ diff,
            const float* __restrict__ icov, float* __restrict__ dist01) {
  const int n    = blockIdx.x;
  const int w    = blockIdx.y;      // 0: N-tiles 0..3, 1: N-tiles 4..6
  const int lane = threadIdx.x;
  const int half = lane >> 4;
  const int half16 = half * 16;
  const int l15  = lane & 15;

  // A fragments (diff already bf16-paired in memory -> raw dword loads)
  BF16x16 a[2][4];
  const unsigned int* dbase =
      (const unsigned int*)(diff + (size_t)n * B_SZ * DPAD2);
#pragma unroll
  for (int mt = 0; mt < 2; ++mt) {
    const unsigned int* row = dbase + (size_t)(mt * 16 + l15) * (DPAD2 / 2);
#pragma unroll
    for (int ks = 0; ks < 4; ++ks) {
#pragma unroll
      for (int j = 0; j < 8; ++j) {
        const int k = ks * 32 + half * 16 + 2 * j;
        a[mt][ks].u[j] = row[k >> 1];
      }
    }
  }

  const float* S = icov + (size_t)n * DPROJ * DPROJ;
  const unsigned short* drows = diff + (size_t)n * B_SZ * DPAD2;
  float part[2][8];
#pragma unroll
  for (int mt = 0; mt < 2; ++mt)
#pragma unroll
    for (int r = 0; r < 8; ++r) part[mt][r] = 0.f;

  if (w == 0) {
    maha_tile<0>(S, drows, a, half16, l15, part);
    maha_tile<1>(S, drows, a, half16, l15, part);
    maha_tile<2>(S, drows, a, half16, l15, part);
    maha_tile<3>(S, drows, a, half16, l15, part);
  } else {
    maha_tile<4>(S, drows, a, half16, l15, part);
    maha_tile<5>(S, drows, a, half16, l15, part);
    maha_tile<6>(S, drows, a, half16, l15, part);
  }

  // reduce across the 16 N-lanes within each half-wave
#pragma unroll
  for (int mt = 0; mt < 2; ++mt)
#pragma unroll
    for (int r = 0; r < 8; ++r) {
      float p = part[mt][r];
      p += __shfl_xor(p, 1, 32);
      p += __shfl_xor(p, 2, 32);
      p += __shfl_xor(p, 4, 32);
      p += __shfl_xor(p, 8, 32);
      part[mt][r] = p;
    }
  float* dpart = dist01 + (size_t)w * B_SZ * NPOS;   // private slice per half
  if (l15 == 0) {
#pragma unroll
    for (int mt = 0; mt < 2; ++mt)
#pragma unroll
      for (int r = 0; r < 8; ++r) {
        const int bb = mt * 16 + r + 8 * half;
        dpart[(size_t)bb * NPOS + n] = part[mt][r];
      }
  }
}

// ---------------------------------------------------------------------------
// Kernel D: sum the two dist halves, normalize, bilinear 64->256
// ---------------------------------------------------------------------------
__global__ void __launch_bounds__(256)
out_upsample(const float* __restrict__ dist01, const float* __restrict__ nmin,
             const float* __restrict__ nmax, float* __restrict__ out) {
  const int idx = blockIdx.x * 256 + threadIdx.x;
  const int b  = idx >> 16;
  const int p  = idx & 65535;
  const int oy = p >> 8, ox = p & 255;
  const float sy = oy * 0.25f - 0.375f;
  const float sx = ox * 0.25f - 0.375f;
  int y0 = (int)floorf(sy); const float fy = sy - (float)y0;
  int x0 = (int)floorf(sx); const float fx = sx - (float)x0;
  const int y1 = min(y0 + 1, HW_FT - 1); y0 = max(y0, 0);
  const int x1 = min(x0 + 1, HW_FT - 1); x0 = max(x0, 0);
  const float* p0 = dist01 + (size_t)b * NPOS;
  const float* p1 = p0 + (size_t)B_SZ * NPOS;
  const float s00 = p0[y0 * 64 + x0] + p1[y0 * 64 + x0];
  const float s01 = p0[y0 * 64 + x1] + p1[y0 * 64 + x1];
  const float s10 = p0[y1 * 64 + x0] + p1[y1 * 64 + x0];
  const float s11 = p0[y1 * 64 + x1] + p1[y1 * 64 + x1];
  const float top = s00 + fx * (s01 - s00);
  const float bot = s10 + fx * (s11 - s10);
  const float v  = top + fy * (bot - top);
  const float mn = nmin[0];
  const float rs = 1.0f / (nmax[0] - mn + 1e-8f);
  out[idx] = (v - mn) * rs;
}

// ---------------------------------------------------------------------------
extern "C" void kernel_launch(void* const* d_in, const int* in_sizes, int n_in,
                              void* d_out, int out_size, void* d_ws, size_t ws_size,
                              hipStream_t stream) {
  (void)in_sizes; (void)n_in; (void)out_size; (void)ws_size;
  const float* comb = (const float*)d_in[0];   // [32,448,32,32]
  const float* pw   = (const float*)d_in[1];   // [100,448]
  const float* pb   = (const float*)d_in[2];   // [100]
  const float* mean = (const float*)d_in[3];   // [100,4096]
  const float* icov = (const float*)d_in[4];   // [4096,100,100]
  const float* nmin = (const float*)d_in[5];   // scalar
  const float* nmax = (const float*)d_in[6];   // scalar
  float* out = (float*)d_out;                  // [32,1,256,256]

  // Workspace layout (all 256B aligned), total ~49 MB
  char* ws = (char*)d_ws;
  float*          proj   = (float*)(ws);                       // 32*1024*112*4 = 14,680,064
  unsigned short* diff   = (unsigned short*)(ws + 14680064);   // 4096*32*128*2 = 33,554,432
  float*          adjT   = (float*)(ws + 14680064 + 33554432); // 4096*128*4    =  2,097,152
  float*          dist01 = (float*)(ws + 14680064 + 33554432 + 2097152); // 2*32*4096*4 = 1,048,576

  make_adj     <<<dim3(NPOS),        dim3(128), 0, stream>>>(mean, pb, adjT);
  proj_gemm    <<<dim3(64, B_SZ),    dim3(32),  0, stream>>>(comb, pw, proj);
  upsample_diff<<<dim3(NPOS, B_SZ),  dim3(128), 0, stream>>>(proj, adjT, diff);
  mahalanobis  <<<dim3(NPOS, 2),     dim3(32),  0, stream>>>(diff, icov, dist01);
  out_upsample <<<dim3(8192),        dim3(256), 0, stream>>>(dist01, nmin, nmax, out);
}
